// BottAttention_38070590112450
// MI455X (gfx1250) — matrix operand order repaired and checked by reading, not compiled, not run on hardware
//
#include <hip/hip_runtime.h>

// ---------------------------------------------------------------------------
// MI455X (gfx1250) attention with post-softmax bias, bf16 WMMA pipeline.
//   out = ( softmax(Q K^T * scale) @ V + reg @ V ) @ W_proj + b_proj
// v_wmma_f32_16x16x32_bf16 everywhere; flash attention stages K/V tiles in
// LDS via async-load-to-LDS + Tensor Data Mover (double buffered).
// GEMM inner loops are ping-pong software-pipelined, unrolled 4x so register
// copies only occur at the (rare) loop backedge.
// ---------------------------------------------------------------------------

#define BATCH 8
#define HEADS 16
#define NTOK 1024
#define HDIM 64
#define CDIM 1024
#define QKVC 3072
#define SCALE 0.125f

typedef __attribute__((ext_vector_type(16))) __bf16 bf16x16;
typedef __attribute__((ext_vector_type(8)))  __bf16 bf16x8;
typedef __attribute__((ext_vector_type(8)))  float  f32x8;
typedef __attribute__((ext_vector_type(4)))  unsigned int u32x4;
typedef __attribute__((ext_vector_type(8)))  int i32x8;
typedef __attribute__((ext_vector_type(4)))  int i32x4;

typedef __attribute__((address_space(1))) i32x4* gv4i_p;   // global int4*
typedef __attribute__((address_space(3))) i32x4* lv4i_p;   // LDS int4*

#if __has_builtin(__builtin_amdgcn_global_load_async_to_lds_b128) && \
    __has_builtin(__builtin_amdgcn_s_wait_asynccnt)
#define USE_ASYNC 1
#else
#define USE_ASYNC 0
#endif

#if __has_builtin(__builtin_amdgcn_tensor_load_to_lds) && \
    __has_builtin(__builtin_amdgcn_s_wait_tensorcnt)
#define USE_TDM 1
#else
#define USE_TDM 0
#endif

__device__ __forceinline__ __bf16 f2bf(float f) {
    union { unsigned short u; __bf16 b; } cv;
    unsigned int x = __float_as_uint(f);
    cv.u = (unsigned short)((x + 0x7FFFu + ((x >> 16) & 1u)) >> 16); // RNE
    return cv.b;
}

__device__ __forceinline__ bf16x16 cat8(bf16x8 lo, bf16x8 hi) {
    return __builtin_shufflevector(lo, hi, 0,1,2,3,4,5,6,7,8,9,10,11,12,13,14,15);
}

__device__ __forceinline__ f32x8 wmma_bf16(bf16x16 a, bf16x16 b, f32x8 c) {
    return __builtin_amdgcn_wmma_f32_16x16x32_bf16(false, a, false, b, (short)0, c, false, false);
}

__device__ __forceinline__ f32x8 fzero() {
    f32x8 z = {0.f,0.f,0.f,0.f,0.f,0.f,0.f,0.f};
    return z;
}

// A-fragment (16x32 bf16, MxK) from row-major [*, ldk]: lane l holds row l%16,
// contiguous 16-elem K chunk at k0 = (l<16?0:16).
__device__ __forceinline__ bf16x16 load_a_frag(const __bf16* base, int ldk, int lane) {
    int row = lane & 15;
    int k0  = (lane < 16) ? 0 : 16;
    const __bf16* p = base + (size_t)row * ldk + k0;
    return cat8(*(const bf16x8*)(p), *(const bf16x8*)(p + 8));
}

// B-fragment (32x16, KxN) with B = M_rowmajor^T: lane l holds column l%16 ==
// row l%16 of M, contiguous 16-elem K chunk at kbase = (l<16?0:16).
__device__ __forceinline__ bf16x16 load_bt_frag(const __bf16* base, int ldk, int lane) {
    int col = lane & 15;
    int kb  = (lane < 16) ? 0 : 16;
    return *(const bf16x16*)(base + (size_t)col * ldk + kb);
}

// 16-byte global->LDS copy: async if available.
__device__ __forceinline__ void copy16_g2l(const void* g, void* l) {
#if USE_ASYNC
    __builtin_amdgcn_global_load_async_to_lds_b128((gv4i_p)g, (lv4i_p)l, 0, 0);
#else
    *(uint4*)l = *(const uint4*)g;
#endif
}

__device__ __forceinline__ void wait_async_copies() {
#if USE_ASYNC
    __builtin_amdgcn_s_wait_asynccnt(0);
#endif
}

// ------------------------------ converts -----------------------------------

__global__ __launch_bounds__(256) void cvt_f32_bf16(const float* __restrict__ in,
                                                    __bf16* __restrict__ out, int n) {
    int i = blockIdx.x * 256 + threadIdx.x;
    if (i < n) out[i] = f2bf(in[i]);
}

__global__ __launch_bounds__(256) void cvt_transpose(const float* __restrict__ in,
                                                     __bf16* __restrict__ out,
                                                     int rows, int cols) {
    int i = blockIdx.x * 256 + threadIdx.x;
    if (i >= rows * cols) return;
    int n = i / rows, k = i % rows;
    out[(size_t)n * rows + k] = f2bf(in[(size_t)k * cols + n]);
}

// ------------------------------ QKV GEMM -----------------------------------
__global__ __launch_bounds__(256) void qkv_gemm(const __bf16* __restrict__ xbf,
                                                const __bf16* __restrict__ wqkvT,
                                                __bf16* __restrict__ qb,
                                                __bf16* __restrict__ kb,
                                                __bf16* __restrict__ vT) {
    int lane = threadIdx.x & 31;
    int wid  = blockIdx.x * 8 + (threadIdx.x >> 5);
    int rt = wid / (QKVC / 64);
    int ct = wid % (QKVC / 64);
    f32x8 acc[4] = {fzero(), fzero(), fzero(), fzero()};

    const __bf16* abase = xbf + (size_t)(rt * 16) * CDIM;
    const __bf16* bbase = wqkvT + (size_t)(ct * 64) * CDIM;

    // ping-pong pipeline: even k-steps use (a0,b0), odd use (a1,b1)
    bf16x16 a0 = load_a_frag(abase, CDIM, lane), a1;
    bf16x16 b0[4], b1[4];
    #pragma unroll
    for (int t = 0; t < 4; ++t)
        b0[t] = load_bt_frag(bbase + (size_t)(t * 16) * CDIM, CDIM, lane);

    #pragma unroll 4
    for (int kt = 0; kt < CDIM / 32; kt += 2) {
        a1 = load_a_frag(abase + (kt + 1) * 32, CDIM, lane);
        #pragma unroll
        for (int t = 0; t < 4; ++t)
            b1[t] = load_bt_frag(bbase + (size_t)(t * 16) * CDIM + (kt + 1) * 32, CDIM, lane);
        #pragma unroll
        for (int t = 0; t < 4; ++t) acc[t] = wmma_bf16(a0, b0[t], acc[t]);

        int k2 = (kt + 2) & (CDIM / 32 - 1);
        a0 = load_a_frag(abase + k2 * 32, CDIM, lane);
        #pragma unroll
        for (int t = 0; t < 4; ++t)
            b0[t] = load_bt_frag(bbase + (size_t)(t * 16) * CDIM + k2 * 32, CDIM, lane);
        #pragma unroll
        for (int t = 0; t < 4; ++t) acc[t] = wmma_bf16(a1, b1[t], acc[t]);
    }
    #pragma unroll
    for (int t = 0; t < 4; ++t) {
        int j = ct * 64 + t * 16 + (lane & 15);
        int three = j >> 10, h = (j >> 6) & 15, d = j & 63;
        #pragma unroll
        for (int v = 0; v < 8; ++v) {
            int M   = (lane < 16) ? v : 8 + v;
            int row = rt * 16 + M;
            int b   = row >> 10, n = row & 1023;
            float val = acc[t][v];
            size_t bh = (size_t)(b * HEADS + h);
            if (three == 0)      qb[(bh * NTOK + n) * HDIM + d] = f2bf(val * SCALE);
            else if (three == 1) kb[(bh * NTOK + n) * HDIM + d] = f2bf(val);
            else                 vT[(bh * HDIM + d) * NTOK + n] = f2bf(val);
        }
    }
}

// ------------------------------ reg @ V ------------------------------------
__global__ __launch_bounds__(256) void regv_gemm(const float* __restrict__ reg,
                                                 const __bf16* __restrict__ vT,
                                                 float* __restrict__ Rv) {
    int lane = threadIdx.x & 31;
    int wid  = blockIdx.x * 8 + (threadIdx.x >> 5);
    int h  = wid / (64 * BATCH);
    int nt = (wid / BATCH) % 64;
    int b  = wid % BATCH;
    f32x8 acc[4] = {fzero(), fzero(), fzero(), fzero()};

    int arow = nt * 16 + (lane & 15);
    int ak   = (lane < 16) ? 0 : 16;
    const float* abase = reg + ((size_t)h * NTOK + arow) * NTOK + ak;
    const __bf16* vbase = vT + (size_t)(b * HEADS + h) * HDIM * NTOK;

    f32x8 lo0 = *(const f32x8*)(abase);
    f32x8 hi0 = *(const f32x8*)(abase + 8);
    f32x8 lo1, hi1;
    bf16x16 b0[4], b1[4];
    #pragma unroll
    for (int t = 0; t < 4; ++t)
        b0[t] = load_bt_frag(vbase + (size_t)(t * 16) * NTOK, NTOK, lane);

    #pragma unroll 4
    for (int kt = 0; kt < NTOK / 32; kt += 2) {
        lo1 = *(const f32x8*)(abase + (kt + 1) * 32);
        hi1 = *(const f32x8*)(abase + (kt + 1) * 32 + 8);
        #pragma unroll
        for (int t = 0; t < 4; ++t)
            b1[t] = load_bt_frag(vbase + (size_t)(t * 16) * NTOK + (kt + 1) * 32, NTOK, lane);
        {
            bf16x16 a;
            #pragma unroll
            for (int i = 0; i < 8; ++i) { a[i] = f2bf(lo0[i]); a[i + 8] = f2bf(hi0[i]); }
            #pragma unroll
            for (int t = 0; t < 4; ++t) acc[t] = wmma_bf16(a, b0[t], acc[t]);
        }
        int k2 = (kt + 2) & (NTOK / 32 - 1);
        lo0 = *(const f32x8*)(abase + k2 * 32);
        hi0 = *(const f32x8*)(abase + k2 * 32 + 8);
        #pragma unroll
        for (int t = 0; t < 4; ++t)
            b0[t] = load_bt_frag(vbase + (size_t)(t * 16) * NTOK + k2 * 32, NTOK, lane);
        {
            bf16x16 a;
            #pragma unroll
            for (int i = 0; i < 8; ++i) { a[i] = f2bf(lo1[i]); a[i + 8] = f2bf(hi1[i]); }
            #pragma unroll
            for (int t = 0; t < 4; ++t) acc[t] = wmma_bf16(a, b1[t], acc[t]);
        }
    }
    #pragma unroll
    for (int t = 0; t < 4; ++t) {
        int d = t * 16 + (lane & 15);
        #pragma unroll
        for (int v = 0; v < 8; ++v) {
            int M = (lane < 16) ? v : 8 + v;
            int row = nt * 16 + M;
            Rv[((size_t)(b * HEADS + h) * NTOK + row) * HDIM + d] = acc[t][v];
        }
    }
}

// --------------------------- flash attention -------------------------------
// One block per (b, h, 128-row query slab); 8 waves, each owns 16 query rows.
// K (32x64) and V^T (64x32) key-block tiles double-buffered in LDS:
//   K tile  -> global_load_async_to_lds_b128 (256 lanes x 16B, contiguous 4KB)
//   V^T tile-> Tensor Data Mover (wave 0 issues tensor_load_to_lds, 2D D#)
__global__ __launch_bounds__(256) void flash_attn(const __bf16* __restrict__ qb,
                                                  const __bf16* __restrict__ kb,
                                                  const __bf16* __restrict__ vT,
                                                  const float* __restrict__ Rv,
                                                  __bf16* __restrict__ attn_out) {
    __shared__ __bf16 ktile[2][32 * 64];   // [key 32][d 64]
    __shared__ __bf16 vtile[2][64 * 32];   // [d 64][key 32]
    __shared__ __bf16 ptile[8][16 * 32];   // per-wave P transpose bounce

    int tid  = threadIdx.x;
    int lane = tid & 31;
    int wslt = tid >> 5;
    size_t bh = blockIdx.x >> 3;                 // (b*HEADS + h)
    int n0 = (blockIdx.x & 7) * 128 + wslt * 16;
    int b  = (int)(bh >> 4), h = (int)(bh & 15);

    const __bf16* kbbase = kb + bh * NTOK * HDIM;
    const __bf16* vtbase = vT + bh * HDIM * NTOK;

    // --- tile stagers ---
    auto stage = [&](int jp, int buf) {
        int key0 = jp * 32;
        // K tile: contiguous 4KB, one b128 per thread
        copy16_g2l((const char*)(kbbase + key0 * HDIM) + tid * 16,
                   (char*)(ktile[buf]) + tid * 16);
#if USE_TDM
        if (wslt == 0) {
            unsigned long long ga =
                (unsigned long long)(size_t)(const void*)(vtbase + key0);
            unsigned lds_off = (unsigned)(size_t)
                ((__attribute__((address_space(3))) char*)(void*)vtile[buf]);
            u32x4 g0;
            g0[0] = 1u;                                   // count=1, user mode
            g0[1] = lds_off;                              // lds_addr (bytes)
            g0[2] = (unsigned)ga;                         // global_addr lo (bytes)
            g0[3] = (unsigned)((ga >> 32) & 0x1FFFFFFull) | 0x80000000u; // type=2
            // data_size=4B dwords: tile 16x64, tensor row 512 dwords, stride 512
            unsigned td0 = 512u, td1 = 4096u, tl0 = 16u, tl1 = 64u, s0 = 512u;
            i32x8 g1;
            g1[0] = (int)(2u << 16);                      // data_size=4B
            g1[1] = (int)((td0 & 0xFFFFu) << 16);
            g1[2] = (int)(((td0 >> 16) & 0xFFFFu) | ((td1 & 0xFFFFu) << 16));
            g1[3] = (int)(((td1 >> 16) & 0xFFFFu) | ((tl0 & 0xFFFFu) << 16));
            g1[4] = (int)(tl1 & 0xFFFFu);                 // tile_dim1 (tile_dim2=0)
            g1[5] = (int)s0;                              // dim0 stride lo
            g1[6] = 0;
            g1[7] = 0;
            i32x4 z4 = {0, 0, 0, 0};
#if __clang_major__ >= 23
            i32x8 z8 = {0, 0, 0, 0, 0, 0, 0, 0};
            __builtin_amdgcn_tensor_load_to_lds(g0, g1, z4, z4, z8, 0);
#else
            __builtin_amdgcn_tensor_load_to_lds(g0, g1, z4, z4, 0);
#endif
        }
#else
        {   // fallback: per-thread 16B copies of the strided V^T tile
            int r = tid >> 2, c = tid & 3;
            copy16_g2l((const char*)(vtbase + (size_t)r * NTOK + key0) + c * 16,
                       (char*)(vtile[buf]) + r * 64 + c * 16);
        }
#endif
    };
    auto stage_wait = [&]() {
        wait_async_copies();
#if USE_TDM
        if (wslt == 0) __builtin_amdgcn_s_wait_tensorcnt(0);
#endif
        __syncthreads();
    };

    // Q fragments (pre-scaled by SCALE in qkv_gemm)
    const __bf16* qbase = qb + (bh * NTOK + n0) * HDIM;
    bf16x16 a0 = load_a_frag(qbase,      HDIM, lane);
    bf16x16 a1 = load_a_frag(qbase + 32, HDIM, lane);

    float mrun[8], lrun[8];
    #pragma unroll
    for (int v = 0; v < 8; ++v) { mrun[v] = -1e30f; lrun[v] = 0.f; }
    f32x8 o[4] = {fzero(), fzero(), fzero(), fzero()};

    int kbo = (lane < 16) ? 0 : 16;
    int c0 = lane & 15, c1 = 16 + (lane & 15);

    stage(0, 0);
    stage_wait();

    for (int jp = 0; jp < NTOK / 32; ++jp) {
        int cur = jp & 1;
        if (jp + 1 < NTOK / 32) stage(jp + 1, cur ^ 1);

        const __bf16* kt = ktile[cur];
        const __bf16* vt = vtile[cur];
        f32x8 s0 = fzero(), s1 = fzero();
        s0 = wmma_bf16(a0, *(const bf16x16*)(kt + c0 * 64 + kbo),      s0);
        s0 = wmma_bf16(a1, *(const bf16x16*)(kt + c0 * 64 + 32 + kbo), s0);
        s1 = wmma_bf16(a0, *(const bf16x16*)(kt + c1 * 64 + kbo),      s1);
        s1 = wmma_bf16(a1, *(const bf16x16*)(kt + c1 * 64 + 32 + kbo), s1);

        float tmax[8];
        #pragma unroll
        for (int v = 0; v < 8; ++v) tmax[v] = fmaxf(s0[v], s1[v]);
        #pragma unroll
        for (int mask = 1; mask <= 8; mask <<= 1)
            #pragma unroll
            for (int v = 0; v < 8; ++v) tmax[v] = fmaxf(tmax[v], __shfl_xor(tmax[v], mask, 32));

        float alpha[8];
        #pragma unroll
        for (int v = 0; v < 8; ++v) {
            float nm = fmaxf(mrun[v], tmax[v]);
            alpha[v] = __expf(mrun[v] - nm);
            mrun[v]  = nm;
        }
        float p0[8], p1[8], rs[8];
        #pragma unroll
        for (int v = 0; v < 8; ++v) {
            p0[v] = __expf(s0[v] - mrun[v]);
            p1[v] = __expf(s1[v] - mrun[v]);
            rs[v] = p0[v] + p1[v];
        }
        #pragma unroll
        for (int mask = 1; mask <= 8; mask <<= 1)
            #pragma unroll
            for (int v = 0; v < 8; ++v) rs[v] += __shfl_xor(rs[v], mask, 32);
        #pragma unroll
        for (int v = 0; v < 8; ++v) lrun[v] = lrun[v] * alpha[v] + rs[v];
        #pragma unroll
        for (int t = 0; t < 4; ++t)
            #pragma unroll
            for (int v = 0; v < 8; ++v) o[t][v] *= alpha[v];

        // C-layout -> A-layout transpose of P through per-wave LDS tile
        __bf16* tile = ptile[wslt];
        int Mo = (lane < 16) ? 0 : 8, cA = lane & 15;
        #pragma unroll
        for (int v = 0; v < 8; ++v) {
            tile[(Mo + v) * 32 + cA]      = f2bf(p0[v]);
            tile[(Mo + v) * 32 + 16 + cA] = f2bf(p1[v]);
        }
        asm volatile("s_wait_dscnt 0x0" ::: "memory");
        bf16x16 pa = load_a_frag(tile, 32, lane);

        #pragma unroll
        for (int t = 0; t < 4; ++t)
            o[t] = wmma_bf16(pa, *(const bf16x16*)(vt + (t * 16 + (lane & 15)) * 32 + kbo), o[t]);

        stage_wait();
    }

    #pragma unroll
    for (int t = 0; t < 4; ++t) {
        int d = t * 16 + (lane & 15);
        #pragma unroll
        for (int v = 0; v < 8; ++v) {
            int M = (lane < 16) ? v : 8 + v;
            int row = n0 + M;
            float val = o[t][v] / lrun[v] + Rv[(bh * NTOK + row) * HDIM + d];
            attn_out[((size_t)b * NTOK + row) * CDIM + h * HDIM + d] = f2bf(val);
        }
    }
}

// ------------------------------ projection ---------------------------------
__global__ __launch_bounds__(256) void proj_gemm(const __bf16* __restrict__ attn_out,
                                                 const __bf16* __restrict__ wprojT,
                                                 const float* __restrict__ bproj,
                                                 float* __restrict__ out) {
    int lane = threadIdx.x & 31;
    int wid  = blockIdx.x * 8 + (threadIdx.x >> 5);
    int rt = wid / (CDIM / 64);
    int ct = wid % (CDIM / 64);
    f32x8 acc[4] = {fzero(), fzero(), fzero(), fzero()};

    const __bf16* abase = attn_out + (size_t)(rt * 16) * CDIM;
    const __bf16* bbase = wprojT + (size_t)(ct * 64) * CDIM;

    bf16x16 a0 = load_a_frag(abase, CDIM, lane), a1;
    bf16x16 b0[4], b1[4];
    #pragma unroll
    for (int t = 0; t < 4; ++t)
        b0[t] = load_bt_frag(bbase + (size_t)(t * 16) * CDIM, CDIM, lane);

    #pragma unroll 4
    for (int kt = 0; kt < CDIM / 32; kt += 2) {
        a1 = load_a_frag(abase + (kt + 1) * 32, CDIM, lane);
        #pragma unroll
        for (int t = 0; t < 4; ++t)
            b1[t] = load_bt_frag(bbase + (size_t)(t * 16) * CDIM + (kt + 1) * 32, CDIM, lane);
        #pragma unroll
        for (int t = 0; t < 4; ++t) acc[t] = wmma_bf16(a0, b0[t], acc[t]);

        int k2 = (kt + 2) & (CDIM / 32 - 1);
        a0 = load_a_frag(abase + k2 * 32, CDIM, lane);
        #pragma unroll
        for (int t = 0; t < 4; ++t)
            b0[t] = load_bt_frag(bbase + (size_t)(t * 16) * CDIM + k2 * 32, CDIM, lane);
        #pragma unroll
        for (int t = 0; t < 4; ++t) acc[t] = wmma_bf16(a1, b1[t], acc[t]);
    }
    #pragma unroll
    for (int t = 0; t < 4; ++t) {
        int col = ct * 64 + t * 16 + (lane & 15);
        float bias = bproj[col];
        #pragma unroll
        for (int v = 0; v < 8; ++v) {
            int M = (lane < 16) ? v : 8 + v;
            out[(size_t)(rt * 16 + M) * CDIM + col] = acc[t][v] + bias;
        }
    }
}

// ------------------------------ launcher -----------------------------------
extern "C" void kernel_launch(void* const* d_in, const int* in_sizes, int n_in,
                              void* d_out, int out_size, void* d_ws, size_t ws_size,
                              hipStream_t stream) {
    const float* x      = (const float*)d_in[0];
    const float* W_qkv  = (const float*)d_in[1];
    const float* reg    = (const float*)d_in[2];
    const float* W_proj = (const float*)d_in[3];
    const float* b_proj = (const float*)d_in[4];
    float* out = (float*)d_out;

    char* ws = (char*)d_ws;
    size_t off = 0;
    auto carve = [&](size_t bytes) { size_t o = off; off += (bytes + 255) & ~(size_t)255; return o; };
    __bf16* xbf    = (__bf16*)(ws + carve((size_t)BATCH * NTOK * CDIM * 2));
    __bf16* wqkvT  = (__bf16*)(ws + carve((size_t)QKVC * CDIM * 2));
    __bf16* wprojT = (__bf16*)(ws + carve((size_t)CDIM * CDIM * 2));
    __bf16* qb     = (__bf16*)(ws + carve((size_t)BATCH * HEADS * NTOK * HDIM * 2));
    __bf16* kb     = (__bf16*)(ws + carve((size_t)BATCH * HEADS * NTOK * HDIM * 2));
    __bf16* vT     = (__bf16*)(ws + carve((size_t)BATCH * HEADS * HDIM * NTOK * 2));
    __bf16* aout   = (__bf16*)(ws + carve((size_t)BATCH * NTOK * CDIM * 2));
    float*  Rv     = (float*)(ws + carve((size_t)BATCH * HEADS * NTOK * HDIM * 4));

    int nx = BATCH * NTOK * CDIM;
    cvt_f32_bf16<<<(nx + 255) / 256, 256, 0, stream>>>(x, xbf, nx);
    cvt_transpose<<<(CDIM * QKVC + 255) / 256, 256, 0, stream>>>(W_qkv, wqkvT, CDIM, QKVC);
    cvt_transpose<<<(CDIM * CDIM + 255) / 256, 256, 0, stream>>>(W_proj, wprojT, CDIM, CDIM);

    qkv_gemm<<<(512 * 48) / 8, 256, 0, stream>>>(xbf, wqkvT, qb, kb, vT);
    regv_gemm<<<(HEADS * 64 * BATCH) / 8, 256, 0, stream>>>(reg, vT, Rv);
    // one block per (b,h,128-row slab)
    flash_attn<<<BATCH * HEADS * 8, 256, 0, stream>>>(qb, kb, vT, Rv, aout);
    proj_gemm<<<(512 * 16) / 8, 256, 0, stream>>>(aout, wprojT, b_proj, out);
}
